// VQCodebook_10204842295880
// MI455X (gfx1250) — matrix operand order repaired and checked by reading, not compiled, not run on hardware
//
#include <hip/hip_runtime.h>

#define NUM_CODES 8192
#define DIM 512
#define NROWS 32768

typedef __attribute__((ext_vector_type(16))) __bf16 v16bf;
typedef __attribute__((ext_vector_type(8)))  float  v8f;

union FragA { int4 q[2]; v16bf v; };

// Raw LDS byte offset of a __shared__ pointer (generic -> AS3 -> int).
__device__ __forceinline__ unsigned __lds_off(void* p) {
    return (unsigned)(unsigned long long)(__attribute__((address_space(3))) char*)p;
}

// Async-copy 64 B (4 x b128) global -> LDS, per lane. ASYNCcnt-tracked.
// Per CDNA5 ISA, INST_OFFSET applies to both global and LDS addr.
__device__ __forceinline__ void async_copy64(const __bf16* g, __bf16* l) {
    unsigned loff = __lds_off((void*)l);
    unsigned long long ga = (unsigned long long)g;
    asm volatile(
        "global_load_async_to_lds_b128 %0, %1, off\n\t"
        "global_load_async_to_lds_b128 %0, %1, off offset:16\n\t"
        "global_load_async_to_lds_b128 %0, %1, off offset:32\n\t"
        "global_load_async_to_lds_b128 %0, %1, off offset:48"
        :: "v"(loff), "v"(ga) : "memory");
}

__device__ __forceinline__ void wait_async0() {
    asm volatile("s_wait_asynccnt 0x0" ::: "memory");
}

// ---------------------------------------------------------------------------
// Prep: W fp32 -> bf16 (workspace), wnorm[k] = ||W_k||^2
// ---------------------------------------------------------------------------
__global__ __launch_bounds__(256)
void vq_prep_kernel(const float* __restrict__ W, __bf16* __restrict__ Wb,
                    float* __restrict__ wnorm)
{
    const int code = blockIdx.x;
    const int t    = threadIdx.x;
    const float* src = W + (size_t)code * DIM;
    __bf16*      dst = Wb + (size_t)code * DIM;
    float s = 0.f;
#pragma unroll
    for (int r = 0; r < DIM / 256; ++r) {
        int i = r * 256 + t;
        float f = src[i];
        s += f * f;
        dst[i] = (__bf16)f;
    }
    __shared__ __attribute__((aligned(16))) float red[256];
    red[t] = s;
    __syncthreads();
    for (int off = 128; off > 0; off >>= 1) {
        if (t < off) red[t] += red[t + off];
        __syncthreads();
    }
    if (t == 0) wnorm[code] = red[0];
}

// ---------------------------------------------------------------------------
// Main: fused bf16-WMMA scores + running argmin of (||w||^2 - 2 z.w)
//   A = 16(codes)x32(dims) W tile, double-buffered async-to-LDS staging.
//   B = two 32x16 z^T fragment sets per wave (32 z-rows) in VGPRs.
//   A ds-loads software-pipelined 2 steps ahead; each feeds 2 WMMA chains.
// ---------------------------------------------------------------------------
__global__ __launch_bounds__(256, 1)
void vq_argmin_kernel(const float* __restrict__ z,
                      const __bf16* __restrict__ Wb,
                      const float* __restrict__ wnorm,
                      int* __restrict__ idx_out)
{
    __shared__ __attribute__((aligned(16))) __bf16 tileW[2][16 * 520]; // 2 x (16 codes x 512+8)
    __shared__ __attribute__((aligned(16))) float  wnormLds[NUM_CODES];

    const int t    = threadIdx.x;
    const int lane = t & 31;
    const int wave = t >> 5;
    const int n    = lane & 15;          // C column (z-row within a 16-row set)
    const int h    = lane >> 4;          // lane half
    const int blockRow = blockIdx.x * 256;
    const int waveRow0 = blockRow + wave * 32;   // 32 rows per wave

    // Stage wnorm into LDS (reused 512x by every wave).
    for (int i = t; i < NUM_CODES / 4; i += 256)
        ((float4*)wnormLds)[i] = ((const float4*)wnorm)[i];

    // Build B fragments once: set r, step s, element j = bf16(z[row n][s*32 + h*8 + (j&7) + (j>>3)*16]).
    v16bf bfrag0[16], bfrag1[16];
#pragma unroll
    for (int s = 0; s < 16; ++s) {
        v16bf b0, b1;
#pragma unroll
        for (int j = 0; j < 16; ++j) {
            b0[j] = (__bf16)z[(size_t)(waveRow0 + n)      * DIM + s * 32 + h * 8 + (j & 7) + (j >> 3) * 16];
            b1[j] = (__bf16)z[(size_t)(waveRow0 + 16 + n) * DIM + s * 32 + h * 8 + (j & 7) + (j >> 3) * 16];
        }
        bfrag0[s] = b0;
        bfrag1[s] = b1;
    }

    float bestV0 = 3.402823466e38f, bestV1 = 3.402823466e38f;
    int   bestI0 = 0,               bestI1 = 0;

    const int lr = t >> 4;               // coop-load: code row 0..15
    const int lc = (t & 15) * 32;        // coop-load: element offset in row

    const int NT = NUM_CODES / 16;       // 512 K-tiles

    // Prime the pipeline: async-load tile 0 into buffer 0.
    async_copy64(Wb + (size_t)lr * DIM + lc, &tileW[0][lr * 520 + lc]);

    for (int kt = 0; kt < NT; ++kt) {
        const int buf = kt & 1;
        wait_async0();        // my async writes for tile kt (and older) done
        __syncthreads();      // everyone's writes visible; prev reads of buf^1 done
        if (kt + 1 < NT) {    // overlap next tile load with this tile's compute
            async_copy64(Wb + (size_t)((kt + 1) * 16 + lr) * DIM + lc,
                         &tileW[buf ^ 1][lr * 520 + lc]);
        }

        // Issue wnorm loads early; consumed only in the epilogue.
        const float4* wn4 = (const float4*)&wnormLds[kt * 16 + h * 8];
        float4 wa = wn4[0], wb = wn4[1];

        const __bf16* abase = &tileW[buf][n * 520 + h * 8];

        // Software-pipelined A-fragment ring, depth 2.
        FragA afr[3];
#pragma unroll
        for (int p = 0; p < 2; ++p) {
            const __bf16* ap = abase + p * 32;
            afr[p].q[0] = *(const int4*)ap;
            afr[p].q[1] = *(const int4*)(ap + 16);
        }

        v8f acc0 = {0.f, 0.f, 0.f, 0.f, 0.f, 0.f, 0.f, 0.f};
        v8f acc1 = {0.f, 0.f, 0.f, 0.f, 0.f, 0.f, 0.f, 0.f};
#pragma unroll
        for (int s = 0; s < 16; ++s) {
            if (s + 2 < 16) {
                const __bf16* ap = abase + (s + 2) * 32;
                afr[(s + 2) % 3].q[0] = *(const int4*)ap;
                afr[(s + 2) % 3].q[1] = *(const int4*)(ap + 16);
            }
            acc0 = __builtin_amdgcn_wmma_f32_16x16x32_bf16(
                false, afr[s % 3].v, false, bfrag0[s], (short)0, acc0, false, false);
            acc1 = __builtin_amdgcn_wmma_f32_16x16x32_bf16(
                false, afr[s % 3].v, false, bfrag1[s], (short)0, acc1, false, false);
        }

        // score = ||w||^2 - 2*dot ; acc[i] -> code kt*16 + h*8 + i
        float wn[8] = {wa.x, wa.y, wa.z, wa.w, wb.x, wb.y, wb.z, wb.w};
#pragma unroll
        for (int i = 0; i < 8; ++i) {
            const int code = kt * 16 + h * 8 + i;
            float s0 = fmaf(-2.0f, acc0[i], wn[i]);
            if (s0 < bestV0) { bestV0 = s0; bestI0 = code; }
            float s1 = fmaf(-2.0f, acc1[i], wn[i]);
            if (s1 < bestV1) { bestV1 = s1; bestI1 = code; }
        }
    }

    // Lanes l and l+16 hold the same z-row: merge pairs (first-min tie-break).
    {
        float oV = __shfl_xor(bestV0, 16, 32);
        int   oI = __shfl_xor(bestI0, 16, 32);
        if (oV < bestV0 || (oV == bestV0 && oI < bestI0)) { bestV0 = oV; bestI0 = oI; }
    }
    {
        float oV = __shfl_xor(bestV1, 16, 32);
        int   oI = __shfl_xor(bestI1, 16, 32);
        if (oV < bestV1 || (oV == bestV1 && oI < bestI1)) { bestV1 = oV; bestI1 = oI; }
    }
    if (h == 0) {
        idx_out[waveRow0 + n]      = bestI0;
        idx_out[waveRow0 + 16 + n] = bestI1;
    }
}

// ---------------------------------------------------------------------------
// Gather: q_st[n] = W[idx[n]] (fp32), idx as float, per-row sq-error partial
// ---------------------------------------------------------------------------
__global__ __launch_bounds__(256)
void vq_gather_kernel(const float* __restrict__ z, const float* __restrict__ W,
                      const int* __restrict__ idx, float* __restrict__ out_q,
                      float* __restrict__ out_idx, float* __restrict__ part)
{
    const int nrow = blockIdx.x;
    const int t    = threadIdx.x;
    const int k    = idx[nrow];
    const float* wrow = W + (size_t)k * DIM;
    const float* zrow = z + (size_t)nrow * DIM;
    float*       qrow = out_q + (size_t)nrow * DIM;
    float s = 0.f;
#pragma unroll
    for (int r = 0; r < DIM / 256; ++r) {
        int i = r * 256 + t;
        float q = wrow[i];
        qrow[i] = q;
        float d = zrow[i] - q;
        s += d * d;
    }
    __shared__ __attribute__((aligned(16))) float red[256];
    red[t] = s;
    __syncthreads();
    for (int off = 128; off > 0; off >>= 1) {
        if (t < off) red[t] += red[t + off];
        __syncthreads();
    }
    if (t == 0) {
        part[nrow]    = red[0];
        out_idx[nrow] = (float)k;
    }
}

// Deterministic final loss reduction (no float atomics).
__global__ __launch_bounds__(256)
void vq_loss_kernel(const float* __restrict__ part, float* __restrict__ out_loss)
{
    const int t = threadIdx.x;
    float s = 0.f;
    for (int i = t; i < NROWS; i += 256) s += part[i];
    __shared__ __attribute__((aligned(16))) float red[256];
    red[t] = s;
    __syncthreads();
    for (int off = 128; off > 0; off >>= 1) {
        if (t < off) red[t] += red[t + off];
        __syncthreads();
    }
    if (t == 0) out_loss[0] = red[0] * (1.0f / ((float)NROWS * (float)DIM));
}

// ---------------------------------------------------------------------------
extern "C" void kernel_launch(void* const* d_in, const int* in_sizes, int n_in,
                              void* d_out, int out_size, void* d_ws, size_t ws_size,
                              hipStream_t stream)
{
    (void)in_sizes; (void)n_in; (void)out_size; (void)ws_size;

    const float* z = (const float*)d_in[0];   // [32768, 512] f32
    const float* W = (const float*)d_in[1];   // [8192, 512] f32

    float* out      = (float*)d_out;
    float* out_q    = out;                                   // [N, DIM]
    float* out_idx  = out + (size_t)NROWS * DIM;             // [N]
    float* out_loss = out_idx + NROWS;                       // [1]

    char* ws = (char*)d_ws;
    __bf16* Wb    = (__bf16*)ws;                                              // 8 MB
    float*  wnorm = (float*)(ws + (size_t)NUM_CODES * DIM * 2);               // 32 KB
    int*    idxb  = (int*)(ws + (size_t)NUM_CODES * DIM * 2 + NUM_CODES * 4); // 128 KB
    float*  part  = (float*)(ws + (size_t)NUM_CODES * DIM * 2 + NUM_CODES * 4
                                + (size_t)NROWS * 4);                         // 128 KB

    vq_prep_kernel<<<NUM_CODES, 256, 0, stream>>>(W, Wb, wnorm);
    vq_argmin_kernel<<<NROWS / 256, 256, 0, stream>>>(z, Wb, wnorm, idxb);
    vq_gather_kernel<<<NROWS, 256, 0, stream>>>(z, W, idxb, out_q, out_idx, part);
    vq_loss_kernel<<<1, 256, 0, stream>>>(part, out_loss);
}
